// MixtureOfExperts_32384053412033
// MI455X (gfx1250) — matrix operand order repaired and checked by reading, not compiled, not run on hardware
//
#include <hip/hip_runtime.h>
#include <hip/hip_bf16.h>
#include <math.h>

#define D 1024
#define H 512
#define E 8
#define T 16384
#define TM 64     // tokens per workgroup tile
#define HS 520    // hlds row stride in bf16 elems (1040B -> 4-bank shift per row)

typedef __attribute__((ext_vector_type(16))) __bf16 bf16x16;
typedef __attribute__((ext_vector_type(8)))  __bf16 bf16x8;
typedef __attribute__((ext_vector_type(8)))  float  f32x8;

union V16 { bf16x16 v; bf16x8 h[2]; };

// Build a 16-wide bf16 fragment from two 16B chunks.
__device__ __forceinline__ bf16x16 ld_frag(const __bf16* p0, const __bf16* p1) {
  V16 u;
  u.h[0] = *(const bf16x8*)p0;
  u.h[1] = *(const bf16x8*)p1;
  return u.v;
}

__device__ __forceinline__ f32x8 wmma_bf16(bf16x16 a, bf16x16 b, f32x8 c) {
  return __builtin_amdgcn_wmma_f32_16x16x32_bf16(
      /*neg_a=*/false, a, /*neg_b=*/false, b,
      /*c_mod=*/(short)0, c, /*reuse_a=*/false, /*reuse_b=*/false);
}

// ---------------------------------------------------------------------------
// Zero output + expert counters
__global__ void zero_init(float* __restrict__ out, int n, int* __restrict__ cnt) {
  int i = blockIdx.x * 256 + threadIdx.x;
  if (i < n) out[i] = 0.0f;
  if (i < E) cnt[i] = 0;
}

// f32 -> bf16 elementwise
__global__ void cvt_bf16(const float* __restrict__ src, __bf16* __restrict__ dst, int n) {
  int i = blockIdx.x * 256 + threadIdx.x;
  if (i < n) dst[i] = (__bf16)src[i];
}

// src: [E][R][C] f32  ->  dst: [E][C][R] bf16   (LDS-tiled transpose)
__global__ void transpose_cvt(const float* __restrict__ src, __bf16* __restrict__ dst,
                              int R, int C) {
  __shared__ float tilebuf[32][33];
  int e = blockIdx.z;
  const float* s = src + (size_t)e * R * C;
  __bf16*      d = dst + (size_t)e * R * C;
  int c0 = blockIdx.x * 32, r0 = blockIdx.y * 32;
  int tx = threadIdx.x, ty = threadIdx.y;
  #pragma unroll
  for (int i = 0; i < 4; ++i)
    tilebuf[ty + 8 * i][tx] = s[(size_t)(r0 + ty + 8 * i) * C + c0 + tx];
  __syncthreads();
  #pragma unroll
  for (int i = 0; i < 4; ++i)
    d[(size_t)(c0 + ty + 8 * i) * R + r0 + tx] = (__bf16)tilebuf[tx][ty + 8 * i];
}

// ---------------------------------------------------------------------------
// Router: one wave per token. gates = x@Wg + bg, top-2, softmax over top-2,
// push (token, weight) into per-expert lists.
__global__ void router(const float* __restrict__ x, const float* __restrict__ Wg,
                       const float* __restrict__ bg, int* __restrict__ cnt,
                       int* __restrict__ tokIdx, float* __restrict__ tokW) {
  int t = blockIdx.x * 8 + (threadIdx.x >> 5);
  int lane = threadIdx.x & 31;
  const float* xr = x + (size_t)t * D;
  float acc[E] = {};
  for (int i = 0; i < D / 32; ++i) {
    int dd = lane + i * 32;
    float xv = xr[dd];
    const float* wr = Wg + dd * E;
    #pragma unroll
    for (int e = 0; e < E; ++e) acc[e] += xv * wr[e];
  }
  #pragma unroll
  for (int off = 16; off > 0; off >>= 1) {
    #pragma unroll
    for (int e = 0; e < E; ++e) acc[e] += __shfl_xor(acc[e], off, 32);
  }
  if (lane == 0) {
    float g[E];
    #pragma unroll
    for (int e = 0; e < E; ++e) g[e] = acc[e] + bg[e];
    int i0 = 0;
    #pragma unroll
    for (int e = 1; e < E; ++e) if (g[e] > g[i0]) i0 = e;
    int i1 = (i0 == 0) ? 1 : 0;
    #pragma unroll
    for (int e = 0; e < E; ++e) if (e != i0 && g[e] > g[i1]) i1 = e;
    float e0 = 1.0f;                      // exp(g[i0]-g[i0])
    float e1 = expf(g[i1] - g[i0]);
    float inv = 1.0f / (e0 + e1);
    int p0 = atomicAdd(&cnt[i0], 1);
    tokIdx[i0 * T + p0] = t; tokW[i0 * T + p0] = e0 * inv;
    int p1 = atomicAdd(&cnt[i1], 1);
    tokIdx[i1 * T + p1] = t; tokW[i1 * T + p1] = e1 * inv;
  }
}

// ---------------------------------------------------------------------------
// Fused expert: 64-token tile, 512 threads = 16 waves arranged (mw 0..3, nw 0..3).
// Stage 1: h[64][512] = gelu(X @ W1 + b1)  (wave: rows 16mw.., cols 128nw..)
// Stage 2: y[64][1024] = h @ W2 (in accumulators; wave: rows 16mw.., cols 256nw..)
// Epilogue: y += b2 + x; LayerNorm via half-wave shuffles + LDS partials;
//           out += w * (LN(y)*gamma + beta)  (f32 atomics; each out elem gets 2 adds)
__global__ void moe_expert(const __bf16* __restrict__ xbf,
                           const float*  __restrict__ xf,
                           const __bf16* __restrict__ W1t,   // [E][H][D]
                           const __bf16* __restrict__ W2t,   // [E][D][H]
                           const float*  __restrict__ b1,
                           const float*  __restrict__ b2,
                           const float*  __restrict__ gamma,
                           const float*  __restrict__ beta,
                           const int*    __restrict__ cnt,
                           const int*    __restrict__ tokIdx,
                           const float*  __restrict__ tokW,
                           float* __restrict__ out) {
  __shared__ __align__(16) __bf16 hlds[TM * HS];   // 66.5 KB
  __shared__ float psum[4][TM];
  __shared__ float psq[4][TM];
  __shared__ float rmu[TM];
  __shared__ float rinv[TM];
  __shared__ int   toks[TM];
  __shared__ float wts[TM];

  const int e = blockIdx.y;
  const int n = cnt[e];
  const int tile = blockIdx.x;
  if (tile * TM >= n) return;

  const int tid  = threadIdx.x;
  const int wave = tid >> 5;
  const int mw   = wave >> 2;     // row group 0..3
  const int nw   = wave & 3;      // col group 0..3
  const int lane = tid & 31;
  const int half = lane >> 4;     // ISA: lanes 0-15 vs 16-31 hold different K chunks / rows
  const int l16  = lane & 15;

  if (tid < TM) {
    int idx = tile * TM + tid;
    bool valid = idx < n;
    if (!valid) idx = n - 1;      // pad with last token, zero weight
    toks[tid] = tokIdx[e * T + idx];
    wts[tid]  = valid ? tokW[e * T + idx] : 0.0f;
  }
  __syncthreads();

  // ---- stage 1: wave owns rows [mw*16,+16) x cols [nw*128,+128) of h
  const __bf16* W1e  = W1t + (size_t)e * H * D;
  const __bf16* arow = xbf + (size_t)toks[mw * 16 + l16] * D + half * 8;
  f32x8 acc1[8] = {};
  for (int k0 = 0; k0 < D; k0 += 32) {
    bf16x16 a = ld_frag(arow + k0, arow + k0 + 16);  // K chunks [0..7|8..15] & [16..23|24..31]
    #pragma unroll
    for (int nt = 0; nt < 8; ++nt) {
      int col = nw * 128 + nt * 16 + l16;
      const __bf16* bp = W1e + (size_t)col * D + k0 + half * 8;  // lanes>=16: K 8..15 & 24..31
      bf16x16 b = ld_frag(bp, bp + 16);
      acc1[nt] = wmma_bf16(a, b, acc1[nt]);
    }
  }
  #pragma unroll
  for (int nt = 0; nt < 8; ++nt) {
    int col = nw * 128 + nt * 16 + l16;
    float bias = b1[e * H + col];
    #pragma unroll
    for (int j = 0; j < 8; ++j) {
      int m = mw * 16 + j + half * 8;          // C/D layout: VGPR j -> rows j / j+8
      float v = acc1[nt][j] + bias;
      float g = 0.5f * v * (1.0f + erff(v * 0.70710678118f));   // exact gelu
      hlds[m * HS + col] = (__bf16)g;
    }
  }
  __syncthreads();

  // ---- stage 2: wave owns rows [mw*16,+16) x cols [nw*256,+256) of y
  const __bf16* W2e  = W2t + (size_t)e * D * H;
  const __bf16* hrow = hlds + (mw * 16 + l16) * HS + half * 8;
  f32x8 acc2[16] = {};
  for (int k0 = 0; k0 < H; k0 += 32) {
    bf16x16 a = ld_frag(hrow + k0, hrow + k0 + 16);
    #pragma unroll
    for (int nt = 0; nt < 16; ++nt) {
      int col = nw * 256 + nt * 16 + l16;
      const __bf16* bp = W2e + (size_t)col * H + k0 + half * 8;
      bf16x16 b = ld_frag(bp, bp + 16);
      acc2[nt] = wmma_bf16(a, b, acc2[nt]);
    }
  }

  // ---- epilogue: residual + b2 folded into accumulators, per-row stats
  float ssum[8], ssq[8];
  #pragma unroll
  for (int j = 0; j < 8; ++j) { ssum[j] = 0.0f; ssq[j] = 0.0f; }
  #pragma unroll
  for (int nt = 0; nt < 16; ++nt) {
    int col = nw * 256 + nt * 16 + l16;
    float bb = b2[e * D + col];
    #pragma unroll
    for (int j = 0; j < 8; ++j) {
      int row = mw * 16 + j + half * 8;
      float v = acc2[nt][j] + bb + xf[(size_t)toks[row] * D + col];
      acc2[nt][j] = v;
      ssum[j] += v;
      ssq[j]  += v * v;
    }
  }
  // reduce the 16 lanes of each half (rows live per-half in the C layout)
  #pragma unroll
  for (int j = 0; j < 8; ++j) {
    #pragma unroll
    for (int off = 8; off > 0; off >>= 1) {
      ssum[j] += __shfl_xor(ssum[j], off, 16);
      ssq[j]  += __shfl_xor(ssq[j],  off, 16);
    }
  }
  if (l16 == 0) {
    #pragma unroll
    for (int j = 0; j < 8; ++j) {
      int row = mw * 16 + j + half * 8;
      psum[nw][row] = ssum[j];
      psq[nw][row]  = ssq[j];
    }
  }
  __syncthreads();
  if (tid < TM) {
    float s = psum[0][tid] + psum[1][tid] + psum[2][tid] + psum[3][tid];
    float q = psq[0][tid] + psq[1][tid] + psq[2][tid] + psq[3][tid];
    float mu  = s * (1.0f / D);
    float var = q * (1.0f / D) - mu * mu;
    rmu[tid]  = mu;
    rinv[tid] = rsqrtf(var + 1e-5f);
  }
  __syncthreads();

  // ---- normalize in-register and combine into out
  #pragma unroll
  for (int nt = 0; nt < 16; ++nt) {
    int col = nw * 256 + nt * 16 + l16;
    float ga = gamma[e * D + col];
    float be = beta[e * D + col];
    #pragma unroll
    for (int j = 0; j < 8; ++j) {
      int row = mw * 16 + j + half * 8;
      float yn = (acc2[nt][j] - rmu[row]) * rinv[row] * ga + be;
      atomicAdd(out + (size_t)toks[row] * D + col, wts[row] * yn);
    }
  }
}

// ---------------------------------------------------------------------------
extern "C" void kernel_launch(void* const* d_in, const int* in_sizes, int n_in,
                              void* d_out, int out_size, void* d_ws, size_t ws_size,
                              hipStream_t stream) {
  (void)in_sizes; (void)n_in; (void)out_size; (void)ws_size;
  const float* x     = (const float*)d_in[0];   // [4,4096,D]
  const float* Wg    = (const float*)d_in[1];   // [D,E]
  const float* bg    = (const float*)d_in[2];   // [E]
  const float* W1    = (const float*)d_in[3];   // [E,D,H]
  const float* b1    = (const float*)d_in[4];   // [E,H]
  const float* W2    = (const float*)d_in[5];   // [E,H,D]
  const float* b2    = (const float*)d_in[6];   // [E,D]
  const float* gamma = (const float*)d_in[7];   // [E,D]
  const float* beta  = (const float*)d_in[8];   // [E,D]
  float* out = (float*)d_out;

  // workspace carve-up (~49.3 MB)
  char* p = (char*)d_ws;
  __bf16* xbf = (__bf16*)p;  p += (size_t)T * D * 2;
  __bf16* W1t = (__bf16*)p;  p += (size_t)E * D * H * 2;
  __bf16* W2t = (__bf16*)p;  p += (size_t)E * D * H * 2;
  int*    cnt = (int*)p;     p += 256;
  int*    tokIdx = (int*)p;  p += (size_t)E * T * 4;
  float*  tokW = (float*)p;  p += (size_t)E * T * 4;

  zero_init<<<(T * D + 255) / 256, 256, 0, stream>>>(out, T * D, cnt);
  cvt_bf16<<<(T * D + 255) / 256, 256, 0, stream>>>(x, xbf, T * D);
  transpose_cvt<<<dim3(H / 32, D / 32, E), dim3(32, 8), 0, stream>>>(W1, W1t, D, H);
  transpose_cvt<<<dim3(D / 32, H / 32, E), dim3(32, 8), 0, stream>>>(W2, W2t, H, D);
  router<<<T / 8, 256, 0, stream>>>(x, Wg, bg, cnt, tokIdx, tokW);
  moe_expert<<<dim3(T / TM, E), 512, 0, stream>>>(xbf, x, W1t, W2t, b1, b2,
                                                  gamma, beta, cnt, tokIdx, tokW, out);
}
